// TransformerTwin_65481071409127
// MI455X (gfx1250) — compile-verified, matrix operand order checked by reading, not run on hardware
//
#include <hip/hip_runtime.h>
#include <hip/hip_bf16.h>

typedef __bf16 bf16_t;
typedef __attribute__((ext_vector_type(16))) __bf16 bf16x16;
typedef __attribute__((ext_vector_type(8)))  __bf16 bf16x8;
typedef __attribute__((ext_vector_type(8)))  float  f32x8;
typedef int vint4 __attribute__((vector_size(16)));

#define NB 8
#define NS 1024
#define ND 768
#define NH 12
#define NDK 64
#define NL 6
#define NF 3072
#define NC 10
#define NM (NB * NS)   // 8192 activation rows

// ---------------------------------------------------------------------------
// CDNA5 async global->LDS copy (ASYNCcnt-tracked), with graceful fallback.
// Builtin signature (from compiler diagnostic): (vint4* src, vint4* lds, Ii, Ii)
// ---------------------------------------------------------------------------
#if defined(__has_builtin)
#if __has_builtin(__builtin_amdgcn_global_load_async_to_lds_b128)
#define HAVE_ASYNC 1
#endif
#endif
#ifndef HAVE_ASYNC
#define HAVE_ASYNC 0
#endif

#if HAVE_ASYNC
#define GPTR(p) ((vint4*)(p))
#define LPTR(p) ((vint4*)(p))
#if __has_builtin(__builtin_amdgcn_s_wait_asynccnt)
#define WAIT_ASYNC(n) __builtin_amdgcn_s_wait_asynccnt(n)
#else
#define WAIT_ASYNC(n) asm volatile("s_wait_asynccnt %0" ::"i"(n))
#endif
#endif

// ---------------------------------------------------------------------------
// WMMA helpers
// ---------------------------------------------------------------------------
__device__ __forceinline__ f32x8 wmma_bf16(bf16x16 a, bf16x16 b, f32x8 c) {
  return __builtin_amdgcn_wmma_f32_16x16x32_bf16(
      false, a, false, b, (short)0, c, false, false);
}

// 16x32 bf16 operand fragment (A layout; B identical when tile is [n][k]).
// Lanes 0-15: row lane&15, K 0..7 / 16..23; lanes 16-31: K 8..15 / 24..31.
__device__ __forceinline__ bf16x16 ldfrag(const bf16_t* base, int stride) {
  int lane = threadIdx.x & 31;
  const bf16_t* p = base + (lane & 15) * stride + ((lane >> 4) << 3);
  bf16x8 lo = *(const bf16x8*)(p);
  bf16x8 hi = *(const bf16x8*)(p + 16);
  bf16x16 f;
#pragma unroll
  for (int i = 0; i < 8; ++i) { f[i] = lo[i]; f[i + 8] = hi[i]; }
  return f;
}

// ---------------------------------------------------------------------------
// DPP16 butterfly reductions over each 16-lane half of the wave (no LDS).
// quad_perm[1,0,3,2]=0xB1 (xor1), quad_perm[2,3,0,1]=0x4E (xor2),
// row_half_mirror=0x141, row_mirror=0x140.
// ---------------------------------------------------------------------------
template <int CTRL>
__device__ __forceinline__ float dpp_mov(float x) {
  return __builtin_bit_cast(
      float, __builtin_amdgcn_update_dpp(0, __builtin_bit_cast(int, x), CTRL,
                                         0xF, 0xF, true));
}
__device__ __forceinline__ float rowmax16(float x) {
  x = fmaxf(x, dpp_mov<0xB1>(x));
  x = fmaxf(x, dpp_mov<0x4E>(x));
  x = fmaxf(x, dpp_mov<0x141>(x));
  x = fmaxf(x, dpp_mov<0x140>(x));
  return x;
}
__device__ __forceinline__ float rowsum16(float x) {
  x += dpp_mov<0xB1>(x);
  x += dpp_mov<0x4E>(x);
  x += dpp_mov<0x141>(x);
  x += dpp_mov<0x140>(x);
  return x;
}

// ---------------------------------------------------------------------------
// bf16 GEMM: C[M,N] = A[M,K] @ B[K,N] with B supplied TRANSPOSED (Bt[N,K]).
// 128x128 macro tile, 8 waves (4x2), double-buffered LDS, async pipeline.
// TRANSV: write bf16 output in [b,h,dk,s] layout (for attention V).
// ---------------------------------------------------------------------------
template <int RELU, int WF32, int WBF16, int TRANSV>
__global__ __launch_bounds__(256) void gemm_bf16_k(
    const bf16_t* __restrict__ A, const bf16_t* __restrict__ Bt,
    float* __restrict__ Cf, bf16_t* __restrict__ Cb, int M, int N, int K) {
  __shared__ bf16_t As[2][128 * 32];  // [m][k]
  __shared__ bf16_t Bs[2][128 * 32];  // [n][k]
  const int tid  = threadIdx.x;
  const int wave = tid >> 5;
  const int wm   = wave >> 1;  // 0..3
  const int wn   = wave & 1;   // 0..1
  const long m0  = (long)blockIdx.x * 128;
  const long n0  = (long)blockIdx.y * 128;
  const int row = tid >> 1, off = (tid & 1) * 16;
  const bf16_t* ga = A  + (m0 + row) * (long)K + off;
  const bf16_t* gb = Bt + (n0 + row) * (long)K + off;

  f32x8 acc[2][4] = {};

  auto stage = [&](int buf, int k0) {
#if HAVE_ASYNC
    __builtin_amdgcn_global_load_async_to_lds_b128(
        GPTR(ga + k0),     LPTR(&As[buf][row * 32 + off]),     0, 0);
    __builtin_amdgcn_global_load_async_to_lds_b128(
        GPTR(ga + k0 + 8), LPTR(&As[buf][row * 32 + off + 8]), 0, 0);
    __builtin_amdgcn_global_load_async_to_lds_b128(
        GPTR(gb + k0),     LPTR(&Bs[buf][row * 32 + off]),     0, 0);
    __builtin_amdgcn_global_load_async_to_lds_b128(
        GPTR(gb + k0 + 8), LPTR(&Bs[buf][row * 32 + off + 8]), 0, 0);
#else
    bf16x8 a0 = *(const bf16x8*)(ga + k0);
    bf16x8 a1 = *(const bf16x8*)(ga + k0 + 8);
    bf16x8 b0 = *(const bf16x8*)(gb + k0);
    bf16x8 b1 = *(const bf16x8*)(gb + k0 + 8);
    *(bf16x8*)(&As[buf][row * 32 + off])     = a0;
    *(bf16x8*)(&As[buf][row * 32 + off + 8]) = a1;
    *(bf16x8*)(&Bs[buf][row * 32 + off])     = b0;
    *(bf16x8*)(&Bs[buf][row * 32 + off + 8]) = b1;
#endif
  };

  const int ntiles = K >> 5;
  stage(0, 0);
  for (int i = 0; i < ntiles; ++i) {
    const int cur = i & 1;
    if (i + 1 < ntiles) {
      stage(cur ^ 1, (i + 1) << 5);
#if HAVE_ASYNC
      WAIT_ASYNC(4);  // own 4 oldest (tile i) have landed
#endif
    } else {
#if HAVE_ASYNC
      WAIT_ASYNC(0);
#endif
    }
    __syncthreads();

    bf16x16 af[2], bfv[4];
#pragma unroll
    for (int ii = 0; ii < 2; ++ii)
      af[ii] = ldfrag(&As[cur][(wm * 32 + ii * 16) * 32], 32);
#pragma unroll
    for (int j = 0; j < 4; ++j)
      bfv[j] = ldfrag(&Bs[cur][(wn * 64 + j * 16) * 32], 32);
#pragma unroll
    for (int ii = 0; ii < 2; ++ii)
#pragma unroll
      for (int j = 0; j < 4; ++j)
        acc[ii][j] = wmma_bf16(af[ii], bfv[j], acc[ii][j]);
    __syncthreads();
  }

  // accumulator: lane holds col (lane&15); VGPR r holds row r + 8*(lane>=16)
  const int lane = tid & 31;
  const int ncol = lane & 15;
  const int mh   = (lane >> 4) * 8;
#pragma unroll
  for (int ii = 0; ii < 2; ++ii)
#pragma unroll
    for (int j = 0; j < 4; ++j)
#pragma unroll
      for (int r = 0; r < 8; ++r) {
        long mrow = m0 + wm * 32 + ii * 16 + mh + r;
        long nc   = n0 + wn * 64 + j * 16 + ncol;
        float v = acc[ii][j][r];
        if (RELU) v = v > 0.f ? v : 0.f;
        if (WF32) Cf[mrow * N + nc] = v;
        if (WBF16) {
          if (TRANSV) {
            // [b, h, dk, s] : b = mrow>>10, s = mrow&1023, h = nc>>6, dk = nc&63
            long idx = (((mrow >> 10) * NH + (nc >> 6)) * NDK + (nc & 63)) * NS +
                       (mrow & (NS - 1));
            Cb[idx] = (bf16_t)v;
          } else {
            Cb[mrow * N + nc] = (bf16_t)v;
          }
        }
      }
}

// ---------------------------------------------------------------------------
// Flash attention: block = (b, h, 64-query tile); 4 waves x 16 query rows.
// K fragments come straight from global [key][dk]; V fragments straight from
// the pre-transposed global layout [b][h][dk][s].  No staging, no barriers.
// ---------------------------------------------------------------------------
__global__ __launch_bounds__(128) void attention_k(
    const bf16_t* __restrict__ Q, const bf16_t* __restrict__ Kg,
    const bf16_t* __restrict__ Vt, bf16_t* __restrict__ O) {
  __shared__ bf16_t Ps[4][16 * 64];  // per-wave P transpose patch [m][key]

  const int tid = threadIdx.x, wave = tid >> 5, lane = tid & 31;
  const int qblk = blockIdx.x, h = blockIdx.y, b = blockIdx.z;
  const long baseRow = (long)b * NS;
  const int  colH = h * NDK;
  const int  q0   = qblk * 64 + wave * 16;

  const bf16_t* qptr  = Q + (baseRow + q0) * ND + colH;
  const bf16_t* kbase = Kg + baseRow * ND + colH;
  const bf16_t* vbase = Vt + (long)(b * NH + h) * NDK * NS;

  bf16x16 qa[2];
#pragma unroll
  for (int ks = 0; ks < 2; ++ks) qa[ks] = ldfrag(qptr + ks * 32, ND);

  f32x8 o[4] = {};
  float mrow[8], lrow[8];
#pragma unroll
  for (int r = 0; r < 8; ++r) { mrow[r] = -1e30f; lrow[r] = 0.f; }

  for (int kb0 = 0; kb0 < NS; kb0 += 64) {
    // S = (Q @ K^T) * 0.125  (16 x 64 per wave)
    f32x8 s[4] = {};
#pragma unroll
    for (int ks = 0; ks < 2; ++ks)
#pragma unroll
      for (int j = 0; j < 4; ++j) {
        bf16x16 kf = ldfrag(kbase + (long)(kb0 + j * 16) * ND + ks * 32, ND);
        s[j] = wmma_bf16(qa[ks], kf, s[j]);
      }

    // online softmax; row stats reduced across 16-lane halves with DPP
#pragma unroll
    for (int r = 0; r < 8; ++r) {
      float mx = -1e30f;
#pragma unroll
      for (int j = 0; j < 4; ++j) {
        float v = s[j][r] * 0.125f;
        s[j][r] = v;
        mx = fmaxf(mx, v);
      }
      mx = rowmax16(mx);
      float mnew  = fmaxf(mrow[r], mx);
      float scale = __expf(mrow[r] - mnew);
      mrow[r] = mnew;
      float rs = 0.f;
#pragma unroll
      for (int j = 0; j < 4; ++j) {
        float p = __expf(s[j][r] - mnew);
        s[j][r] = p;
        rs += p;
      }
      rs = rowsum16(rs);
      lrow[r] = lrow[r] * scale + rs;
#pragma unroll
      for (int j = 0; j < 4; ++j) o[j][r] *= scale;
    }

    // accumulator layout -> A-fragment layout via per-wave LDS patch
    bf16_t* pw = Ps[wave];
    const int nlo = lane & 15, hi8 = (lane >> 4) * 8;
#pragma unroll
    for (int j = 0; j < 4; ++j)
#pragma unroll
      for (int r = 0; r < 8; ++r)
        pw[(hi8 + r) * 64 + j * 16 + nlo] = (bf16_t)s[j][r];

    // O += P @ V
#pragma unroll
    for (int ks = 0; ks < 2; ++ks) {
      bf16x16 pa = ldfrag(pw + ks * 32, 64);
#pragma unroll
      for (int j = 0; j < 4; ++j) {
        bf16x16 vf = ldfrag(vbase + (long)(j * 16) * NS + kb0 + ks * 32, NS);
        o[j] = wmma_bf16(pa, vf, o[j]);
      }
    }
  }

  const int nlo = lane & 15, hi8 = (lane >> 4) * 8;
#pragma unroll
  for (int j = 0; j < 4; ++j)
#pragma unroll
    for (int r = 0; r < 8; ++r) {
      float inv = 1.f / lrow[r];
      O[(baseRow + q0 + hi8 + r) * ND + colH + j * 16 + nlo] =
          (bf16_t)(o[j][r] * inv);
    }
}

// ---------------------------------------------------------------------------
// residual add + LayerNorm (one block per row of 768)
// ---------------------------------------------------------------------------
__global__ __launch_bounds__(256) void add_ln_k(
    const float* __restrict__ xres, const float* __restrict__ acc,
    const float* __restrict__ sc, const float* __restrict__ bi,
    float* __restrict__ xout, bf16_t* __restrict__ xh) {
  __shared__ float red[256];
  const long row = blockIdx.x;
  const int tid = threadIdx.x;
  float v[3];
  float sum = 0.f;
#pragma unroll
  for (int i = 0; i < 3; ++i) {
    int c = tid + i * 256;
    float t = xres[row * ND + c] + acc[row * ND + c];
    v[i] = t;
    sum += t;
  }
  red[tid] = sum;
  __syncthreads();
  for (int st = 128; st > 0; st >>= 1) {
    if (tid < st) red[tid] += red[tid + st];
    __syncthreads();
  }
  float mu = red[0] * (1.f / ND);
  __syncthreads();
  float vs = 0.f;
#pragma unroll
  for (int i = 0; i < 3; ++i) { float d = v[i] - mu; vs += d * d; }
  red[tid] = vs;
  __syncthreads();
  for (int st = 128; st > 0; st >>= 1) {
    if (tid < st) red[tid] += red[tid + st];
    __syncthreads();
  }
  float rstd = rsqrtf(red[0] * (1.f / ND) + 1e-5f);
#pragma unroll
  for (int i = 0; i < 3; ++i) {
    int c = tid + i * 256;
    float y = (v[i] - mu) * rstd * sc[c] + bi[c];
    xout[row * ND + c] = y;
    xh[row * ND + c] = (bf16_t)y;
  }
}

// ---------------------------------------------------------------------------
// small kernels
// ---------------------------------------------------------------------------
__global__ void embed_pe_k(const int* __restrict__ tok,
                           const float* __restrict__ emb,
                           float* __restrict__ x, bf16_t* __restrict__ xh,
                           long total) {
  long i = (long)blockIdx.x * blockDim.x + threadIdx.x;
  if (i >= total) return;
  int d = (int)(i % ND);
  long bs = i / ND;
  int s = (int)(bs % NS);
  int t = tok[bs];
  float div = __expf((float)(d & ~1) * (-9.210340371976184f / (float)ND));
  float ang = (float)s * div;
  float pe = (d & 1) ? __cosf(ang) : __sinf(ang);
  float val = emb[(long)t * ND + d] + pe;
  x[i] = val;
  xh[i] = (bf16_t)val;
}

// fp32 [L][K][N] -> bf16 transposed [L][N][K]
__global__ void f2bf_t_k(const float* __restrict__ in, bf16_t* __restrict__ out,
                         int K, int N, long total) {
  long i = (long)blockIdx.x * blockDim.x + threadIdx.x;
  if (i >= total) return;
  long per = (long)K * N;
  long l = i / per, r = i % per;
  long n = r / K, k = r % K;
  out[i] = (bf16_t)in[l * per + k * N + n];
}

__global__ void pool_k(const float* __restrict__ x, float* __restrict__ pooled) {
  int i = blockIdx.x * blockDim.x + threadIdx.x;  // B*D
  if (i >= NB * ND) return;
  int b = i / ND, d = i % ND;
  float s = 0.f;
  for (int t = 0; t < NS; ++t) s += x[((long)b * NS + t) * ND + d];
  pooled[i] = s * (1.f / NS);
}

__global__ void cls_k(const float* __restrict__ pooled,
                      const float* __restrict__ Wc,
                      const float* __restrict__ bc, float* __restrict__ out) {
  int i = blockIdx.x * blockDim.x + threadIdx.x;  // B*C = 80
  if (i >= NB * NC) return;
  int b = i / NC, c = i % NC;
  float s = bc[c];
  for (int d = 0; d < ND; ++d) s += pooled[b * ND + d] * Wc[d * NC + c];
  out[i] = s;
}

// ---------------------------------------------------------------------------
// host
// ---------------------------------------------------------------------------
extern "C" void kernel_launch(void* const* d_in, const int* in_sizes, int n_in,
                              void* d_out, int out_size, void* d_ws,
                              size_t ws_size, hipStream_t stream) {
  (void)in_sizes; (void)n_in; (void)out_size; (void)ws_size;
  const int*   tokens = (const int*)d_in[0];
  const float* emb    = (const float*)d_in[1];
  const float* Wq     = (const float*)d_in[2];
  const float* Wk     = (const float*)d_in[3];
  const float* Wv     = (const float*)d_in[4];
  const float* Wo     = (const float*)d_in[5];
  const float* W1     = (const float*)d_in[6];
  const float* W2     = (const float*)d_in[7];
  const float* ln1s   = (const float*)d_in[8];
  const float* ln1b   = (const float*)d_in[9];
  const float* ln2s   = (const float*)d_in[10];
  const float* ln2b   = (const float*)d_in[11];
  const float* Wc     = (const float*)d_in[12];
  const float* bc     = (const float*)d_in[13];

  const long DD = (long)ND * ND;
  const long DF = (long)ND * NF;
  const long MD = (long)NM * ND;
  const long MF = (long)NM * NF;

  char* wsp = (char*)d_ws;
  auto alloc = [&](size_t bytes) -> char* {
    char* p = wsp;
    wsp += (bytes + 255) & ~(size_t)255;
    return p;
  };
  bf16_t* wqT   = (bf16_t*)alloc(NL * DD * 2);  // [l][n][k]
  bf16_t* wkT   = (bf16_t*)alloc(NL * DD * 2);
  bf16_t* wvT   = (bf16_t*)alloc(NL * DD * 2);
  bf16_t* woT   = (bf16_t*)alloc(NL * DD * 2);
  bf16_t* w1T   = (bf16_t*)alloc(NL * DF * 2);  // [l][F][D]
  bf16_t* w2T   = (bf16_t*)alloc(NL * DF * 2);  // [l][D][F]
  float*  x     = (float*)alloc(MD * 4);
  bf16_t* xh    = (bf16_t*)alloc(MD * 2);
  bf16_t* qb    = (bf16_t*)alloc(MD * 2);
  bf16_t* kbuf  = (bf16_t*)alloc(MD * 2);
  bf16_t* vt    = (bf16_t*)alloc(MD * 2);       // [b][h][dk][s]
  bf16_t* attnb = (bf16_t*)alloc(MD * 2);
  bf16_t* h1    = (bf16_t*)alloc(MF * 2);
  float*  gacc  = (float*)alloc(MD * 4);
  float*  pooled = (float*)alloc(NB * ND * 4);

  {  // weights -> bf16, transposed to [N][K] per layer
    long nDD = NL * DD, nDF = NL * DF;
    int gDD = (int)((nDD + 255) / 256), gDF = (int)((nDF + 255) / 256);
    f2bf_t_k<<<gDD, 256, 0, stream>>>(Wq, wqT, ND, ND, nDD);
    f2bf_t_k<<<gDD, 256, 0, stream>>>(Wk, wkT, ND, ND, nDD);
    f2bf_t_k<<<gDD, 256, 0, stream>>>(Wv, wvT, ND, ND, nDD);
    f2bf_t_k<<<gDD, 256, 0, stream>>>(Wo, woT, ND, ND, nDD);
    f2bf_t_k<<<gDF, 256, 0, stream>>>(W1, w1T, ND, NF, nDF);
    f2bf_t_k<<<gDF, 256, 0, stream>>>(W2, w2T, NF, ND, nDF);
  }

  embed_pe_k<<<(int)((MD + 255) / 256), 256, 0, stream>>>(tokens, emb, x, xh, MD);

  const dim3 gD(NM / 128, ND / 128);  // 64 x 6
  const dim3 gF(NM / 128, NF / 128);  // 64 x 24
  const dim3 gA(NS / 64, NH, NB);     // 16 x 12 x 8

  for (int l = 0; l < NL; ++l) {
    const bf16_t* wq_l = wqT + (long)l * DD;
    const bf16_t* wk_l = wkT + (long)l * DD;
    const bf16_t* wv_l = wvT + (long)l * DD;
    const bf16_t* wo_l = woT + (long)l * DD;
    const bf16_t* w1_l = w1T + (long)l * DF;
    const bf16_t* w2_l = w2T + (long)l * DF;

    gemm_bf16_k<0, 0, 1, 0><<<gD, 256, 0, stream>>>(xh, wq_l, nullptr, qb, NM, ND, ND);
    gemm_bf16_k<0, 0, 1, 0><<<gD, 256, 0, stream>>>(xh, wk_l, nullptr, kbuf, NM, ND, ND);
    gemm_bf16_k<0, 0, 1, 1><<<gD, 256, 0, stream>>>(xh, wv_l, nullptr, vt, NM, ND, ND);

    attention_k<<<gA, 128, 0, stream>>>(qb, kbuf, vt, attnb);

    gemm_bf16_k<0, 1, 0, 0><<<gD, 256, 0, stream>>>(attnb, wo_l, gacc, nullptr, NM, ND, ND);
    add_ln_k<<<NM, 256, 0, stream>>>(x, gacc, ln1s + (long)l * ND,
                                     ln1b + (long)l * ND, x, xh);

    gemm_bf16_k<1, 0, 1, 0><<<gF, 256, 0, stream>>>(xh, w1_l, nullptr, h1, NM, NF, ND);
    gemm_bf16_k<0, 1, 0, 0><<<gD, 256, 0, stream>>>(h1, w2_l, gacc, nullptr, NM, ND, NF);
    add_ln_k<<<NM, 256, 0, stream>>>(x, gacc, ln2s + (long)l * ND,
                                     ln2b + (long)l * ND, x, xh);
  }

  pool_k<<<(NB * ND + 255) / 256, 256, 0, stream>>>(x, pooled);
  cls_k<<<1, 128, 0, stream>>>(pooled, Wc, bc, (float*)d_out);
}
